// LinearST_82832739270909
// MI455X (gfx1250) — matrix operand order repaired and checked by reading, not compiled, not run on hardware
//
#include <hip/hip_runtime.h>

#define PASTW   12
#define DIMSZ   256
#define SEQLEN  288
#define NBATCH  64
#define NWIN    (SEQLEN - PASTW)        // 276
#define KDIM    (PASTW * DIMSZ)         // 3072
#define NDIM    (PASTW * DIMSZ)         // 3072 output features
#define MROWS   (NBATCH * NWIN)         // 17664

typedef _Float16 v16h __attribute__((ext_vector_type(16)));
typedef _Float16 v8h  __attribute__((ext_vector_type(8)));
typedef _Float16 v4h  __attribute__((ext_vector_type(4)));
typedef float    v8f  __attribute__((ext_vector_type(8)));
typedef float    v4f  __attribute__((ext_vector_type(4)));

// ---------------------------------------------------------------------------
// f32 -> f16 conversion (4 elements / thread, b128 load + b64 store)
// ---------------------------------------------------------------------------
__global__ __launch_bounds__(256) void cvt_f32_to_f16(
    const float* __restrict__ src, _Float16* __restrict__ dst, int n4) {
  int i = blockIdx.x * blockDim.x + threadIdx.x;
  if (i < n4) {
    v4f x = *(const v4f*)(src + 4 * (size_t)i);
    v4h h;
    h.x = (_Float16)x.x;
    h.y = (_Float16)x.y;
    h.z = (_Float16)x.z;
    h.w = (_Float16)x.w;
    *(v4h*)(dst + 4 * (size_t)i) = h;
  }
}

// ---------------------------------------------------------------------------
// WMMA GEMM: out[m, o] = sum_k Ah[m,k] * Wh[o,k] + bias[o]
// Block: 256 threads = 8 waves (2 M x 4 N), block tile 128(M) x 256(N).
// Wave tile: 64(M) x 64(N) = 4x4 accumulator fragments (16 loads / 16 WMMA
// per 32-K step -> balanced vmem vs matrix issue).
// Output uses non-temporal stores so the 217 MB stream does not evict the
// L2-resident f16 W (18.9 MB) + input (9.4 MB) operands.
// ---------------------------------------------------------------------------
__global__ __launch_bounds__(256) void gemm_wmma_f16(
    const _Float16* __restrict__ Ah,    // (NBATCH*SEQLEN*DIMSZ) halfs
    const _Float16* __restrict__ Wh,    // (NDIM, KDIM) halfs, row-major
    const float*    __restrict__ bias,  // (NDIM)
    float*          __restrict__ out)   // (MROWS, NDIM)
{
  const int ntile = blockIdx.x;   // 0..11   (N / 256)
  const int mtile = blockIdx.y;   // 0..137  (M / 128)
  const int tid   = threadIdx.x;
  const int wave  = tid >> 5;
  const int lane  = tid & 31;
  const int wm    = wave & 1;     // M sub-tile (2 x 64 rows)
  const int wn    = wave >> 1;    // N sub-tile (4 x 64 cols)
  const int g     = lane >> 4;    // lane half-group
  const int lm    = lane & 15;

  // Per-lane A row base pointers (ISA A layout: lane L -> M=L%16,
  // halves 0..7 = K 8g+0..7, halves 8..15 = K 16+8g+0..7)
  const _Float16* aRow[4];
#pragma unroll
  for (int fr = 0; fr < 4; ++fr) {
    int m     = mtile * 128 + wm * 64 + fr * 16 + lm;
    int batch = m / NWIN;
    int nwin  = m - batch * NWIN;
    aRow[fr]  = Ah + (size_t)batch * (SEQLEN * DIMSZ) + (size_t)nwin * DIMSZ + 8 * g;
  }

  // Per-lane B row base pointers (B layout: lane = column o, halves 0..7 =
  // K 8g+0..7, halves 8..15 = K 16+8g+0..7 of W row o)
  const _Float16* bRow[4];
  v8f acc[4][4];
#pragma unroll
  for (int fc = 0; fc < 4; ++fc) {
    int o    = ntile * 256 + wn * 64 + fc * 16 + lm;
    bRow[fc] = Wh + (size_t)o * KDIM + 8 * g;
    float bv = bias[o];
#pragma unroll
    for (int fr = 0; fr < 4; ++fr)
#pragma unroll
      for (int e = 0; e < 8; ++e)
        acc[fr][fc][e] = bv;   // bias pre-loaded into accumulator
  }

  // K loop: 96 steps of 32
#pragma unroll 1
  for (int kk = 0; kk < KDIM; kk += 32) {
    v16h afrag[4], bfrag[4];
#pragma unroll
    for (int fr = 0; fr < 4; ++fr) {
      v8h lo = *(const v8h*)(aRow[fr] + kk);        // K = kk+8g .. +7
      v8h hi = *(const v8h*)(aRow[fr] + kk + 16);   // K = kk+16+8g .. +7
      afrag[fr] = __builtin_shufflevector(lo, hi,
          0, 1, 2, 3, 4, 5, 6, 7, 8, 9, 10, 11, 12, 13, 14, 15);
    }
#pragma unroll
    for (int fc = 0; fc < 4; ++fc) {
      v8h lo = *(const v8h*)(bRow[fc] + kk);        // K = kk+8g .. +7
      v8h hi = *(const v8h*)(bRow[fc] + kk + 16);   // K = kk+16+8g .. +7
      bfrag[fc] = __builtin_shufflevector(lo, hi,
          0, 1, 2, 3, 4, 5, 6, 7, 8, 9, 10, 11, 12, 13, 14, 15);
    }
#pragma unroll
    for (int fr = 0; fr < 4; ++fr)
#pragma unroll
      for (int fc = 0; fc < 4; ++fc)
        acc[fr][fc] = __builtin_amdgcn_wmma_f32_16x16x32_f16(
            false, afrag[fr], false, bfrag[fc],
            (short)0, acc[fr][fc], false, false);
  }

  // Store: D element (M = r + 8g, N = lm) -> out[m*NDIM + o], f32, NT hint
#pragma unroll
  for (int fr = 0; fr < 4; ++fr) {
    size_t mBase = (size_t)(mtile * 128 + wm * 64 + fr * 16 + 8 * g);
    float* orow  = out + mBase * NDIM + (ntile * 256 + wn * 64 + lm);
#pragma unroll
    for (int fc = 0; fc < 4; ++fc)
#pragma unroll
      for (int r = 0; r < 8; ++r)
        __builtin_nontemporal_store(acc[fr][fc][r],
                                    orow + (size_t)r * NDIM + fc * 16);
  }
}

// ---------------------------------------------------------------------------
extern "C" void kernel_launch(void* const* d_in, const int* in_sizes, int n_in,
                              void* d_out, int out_size, void* d_ws, size_t ws_size,
                              hipStream_t stream) {
  const float* inp  = (const float*)d_in[0];   // (64, 288, 256) f32
  const float* W    = (const float*)d_in[1];   // (3072, 3072)   f32
  const float* bias = (const float*)d_in[2];   // (3072)         f32
  float* out = (float*)d_out;                  // (17664, 3072)  f32

  // Workspace: Wh (18.87 MB) then Ah (9.44 MB) -> 28.3 MB total
  _Float16* Wh = (_Float16*)d_ws;
  _Float16* Ah = (_Float16*)((char*)d_ws + (size_t)NDIM * KDIM * sizeof(_Float16));

  const int nW4 = (NDIM * KDIM) / 4;                 // 2,359,296
  const int nA4 = (NBATCH * SEQLEN * DIMSZ) / 4;     // 1,179,648
  cvt_f32_to_f16<<<(nW4 + 255) / 256, 256, 0, stream>>>(W, Wh, nW4);
  cvt_f32_to_f16<<<(nA4 + 255) / 256, 256, 0, stream>>>(inp, Ah, nA4);

  dim3 grid(NDIM / 256, MROWS / 128);   // (12, 138) = 1656 workgroups
  gemm_wmma_f16<<<grid, 256, 0, stream>>>(Ah, Wh, bias, out);
}
